// GraphTripleConv_6459630813308
// MI455X (gfx1250) — matrix operand order repaired and checked by reading, not compiled, not run on hardware
//
#include <hip/hip_runtime.h>

typedef __attribute__((ext_vector_type(16))) __bf16 v16bf;
typedef __attribute__((ext_vector_type(8)))  float  v8f;

#define O_N    50000
#define T_N    200000
#define DIN_N  128
#define H_N    256
#define DOUT_N 128

#if defined(__has_builtin)
#if __has_builtin(__builtin_amdgcn_global_load_async_to_lds_b128) && \
    __has_builtin(__builtin_amdgcn_s_wait_asynccnt)
#define USE_ASYNC_LDS 1
#endif
#endif
#ifndef USE_ASYNC_LDS
#define USE_ASYNC_LDS 0
#endif

#if USE_ASYNC_LDS
// Match the builtin's exact parameter types: int __vector(4) with AS1 / AS3.
typedef int v4i_ls __attribute__((vector_size(16)));
typedef __attribute__((address_space(1))) v4i_ls* g_v4p;
typedef __attribute__((address_space(3))) v4i_ls* l_v4p;
#endif

// ---------- helpers ----------

__device__ __forceinline__ unsigned short f2bf(float f) {
  union { float f; unsigned u; } U; U.f = f;
  unsigned r = U.u + 0x7FFFu + ((U.u >> 16) & 1u);   // round-to-nearest-even
  return (unsigned short)(r >> 16);
}

union FragU { uint4 u[2]; v16bf v; };

// A-frag (16x32 bf16) from an LDS row-major tile. Lane L: m = L%16, g = L/16.
// e=0..7 -> K = kk*32 + g*8 + e ; e=8..15 -> K = kk*32 + 16 + g*8 + (e-8)
__device__ __forceinline__ v16bf lds_afrag(const unsigned short* rowbase, int kk, int g) {
  FragU U;
  const unsigned short* p = rowbase + kk * 32 + g * 8;
  U.u[0] = *(const uint4*)(p);
  U.u[1] = *(const uint4*)(p + 16);
  return U.v;
}

__device__ __forceinline__ v8f wmma_bf16(v16bf a, v16bf b, v8f c) {
  return __builtin_amdgcn_wmma_f32_16x16x32_bf16(false, a, false, b, (short)0, c,
                                                 false, false);
}

// One k-tile of the GEMM: NT column-tiles, B-frags streamed from an LDS slice.
template <int NT>
__device__ __forceinline__ void gemm_ktile(const unsigned short* arow, int kk, int g,
                                           int lane, const unsigned short* bslice,
                                           v8f* acc) {
  const unsigned short* bp = bslice + lane * 16;
  v16bf a = lds_afrag(arow, kk, g);
#pragma unroll
  for (int nt = 0; nt < NT; ++nt) {
    v16bf b = *(const v16bf*)(bp + (size_t)nt * 512);
    acc[nt] = wmma_bf16(a, b, acc[nt]);
  }
}

// Cooperatively stage n16 * 16 bytes of packed weights into LDS (all 256 threads).
__device__ __forceinline__ void stage_slice(const unsigned short* __restrict__ src,
                                            unsigned short* dst, int n16, int tid) {
  const uint4* s = (const uint4*)src;
  uint4* d = (uint4*)dst;
#if USE_ASYNC_LDS
  for (int i = tid; i < n16; i += 256) {
    __builtin_amdgcn_global_load_async_to_lds_b128(
        (g_v4p)(unsigned long long)(uintptr_t)(s + i),
        (l_v4p)(unsigned)(uintptr_t)(d + i),
        0, 0);
  }
#else
  for (int i = tid; i < n16; i += 256) d[i] = s[i];
#endif
}

__device__ __forceinline__ void stage_sync() {
#if USE_ASYNC_LDS
  __builtin_amdgcn_s_wait_asynccnt(0);
#endif
  __syncthreads();
}

// ---------- aux kernels ----------

__global__ void zero_f32(float* __restrict__ p, long n) {
  long i = (long)blockIdx.x * blockDim.x + threadIdx.x;
  long stride = (long)gridDim.x * blockDim.x;
  for (; i < n; i += stride) p[i] = 0.0f;
}

// Pack W (Kd x Nd, row-major f32) into bf16 WMMA-B frag layout.
// frag = kk*(Nd/16)+nn ; lane L (g=L/16,c=L%16) ; element e: K = kk*32+(e/8)*16+g*8+(e%8), col = nn*16+c
__global__ void pack_w(const float* __restrict__ W, unsigned short* __restrict__ dst,
                       int Kd, int Nd) {
  int i = blockIdx.x * blockDim.x + threadIdx.x;
  int n = Kd * Nd;
  if (i >= n) return;
  int e = i & 15;
  int L = (i >> 4) & 31;
  int frag = i >> 9;
  int nT = Nd >> 4;
  int nn = frag % nT;
  int kk = frag / nT;
  int krow = kk * 32 + (e >> 3) * 16 + (L >> 4) * 8 + (e & 7);
  int col  = nn * 16 + (L & 15);
  dst[i] = f2bf(W[(size_t)krow * Nd + col]);
}

__global__ void count_edges(const int* __restrict__ edges, float* __restrict__ counts) {
  int t = blockIdx.x * blockDim.x + threadIdx.x;
  if (t >= T_N) return;
  atomicAdd(&counts[edges[2 * t]],     1.0f);
  atomicAdd(&counts[edges[2 * t + 1]], 1.0f);
}

// ---------- fused per-triple MLP ----------
// Block = 8 waves = 128 triples. Waves share double-buffered LDS weight slices.
__global__ __launch_bounds__(256) void triple_mlp(
    const float* __restrict__ obj, const float* __restrict__ pred,
    const int* __restrict__ edges,
    const unsigned short* __restrict__ pw1, const float* __restrict__ b1,
    const unsigned short* __restrict__ pw2, const float* __restrict__ b2,
    float* __restrict__ pooled, float* __restrict__ outP)
{
  __shared__ __align__(16) unsigned short curh[8][16 * 384]; // cur_t bf16; reused for h
  __shared__ __align__(16) unsigned short wbuf[2][20 * 512]; // staged weight slices
  __shared__ int sSh[8][16];
  __shared__ int oSh[8][16];

  const int tid  = threadIdx.x;
  const int wid  = tid >> 5;
  const int lane = tid & 31;
  const int g = lane >> 4, c = lane & 15;
  const int t0 = blockIdx.x * 128 + wid * 16;
  const bool active = (t0 < T_N);   // uniform per wave; inactive waves only stage+barrier

  // ---- gather cur_t = [obj[s], pred, obj[o]] -> bf16 LDS [16][384] ----
  if (active) {
    for (int m = 0; m < 16; ++m) {
      const int t  = t0 + m;
      const int si = edges[2 * t];
      const int oi = edges[2 * t + 1];
      if (lane == 0) { sSh[wid][m] = si; oSh[wid][m] = oi; }
      const float* s0 = obj  + (size_t)si * DIN_N;
      const float* s1 = pred + (size_t)t  * DIN_N;
      const float* s2 = obj  + (size_t)oi * DIN_N;
#pragma unroll
      for (int j = 0; j < 3; ++j) {
        const float* src = (j == 0) ? s0 : (j == 1) ? s1 : s2;
        float4 v = *(const float4*)(src + lane * 4);
        unsigned lo = (unsigned)f2bf(v.x) | ((unsigned)f2bf(v.y) << 16);
        unsigned hi = (unsigned)f2bf(v.z) | ((unsigned)f2bf(v.w) << 16);
        *(uint2*)&curh[wid][m * 384 + j * 128 + lane * 4] = make_uint2(lo, hi);
      }
    }
  }

  // ---- stage 1: h = relu(cur_t @ W1 + b1) ; K=384 (12 ktiles), N=256 (16 ntiles) ----
  v8f acc[16];
  if (active) {
#pragma unroll
    for (int nt = 0; nt < 16; ++nt) {
      float bv = b1[nt * 16 + c];
      acc[nt] = (v8f){bv, bv, bv, bv, bv, bv, bv, bv};
    }
  }
  stage_slice(pw1, wbuf[0], 16 * 64, tid);
  stage_sync();
  for (int kk = 0; kk < 12; ++kk) {
    if (kk + 1 < 12)
      stage_slice(pw1 + (size_t)((kk + 1) * 16) * 512, wbuf[(kk + 1) & 1], 16 * 64, tid);
    if (active)
      gemm_ktile<16>(&curh[wid][c * 384], kk, g, lane, wbuf[kk & 1], acc);
    stage_sync();
  }
  // relu + restage h as bf16 into LDS [16][256] (reuse curh region; wave-private)
  if (active) {
#pragma unroll
    for (int nt = 0; nt < 16; ++nt) {
#pragma unroll
      for (int r = 0; r < 8; ++r) {
        float v = fmaxf(acc[nt][r], 0.0f);
        int m = r + 8 * g;
        curh[wid][m * 256 + nt * 16 + c] = f2bf(v);
      }
    }
  }

  // ---- stage 2: new_t = relu(h @ W2 + b2) ; K=256 (8 kt), N=640 (40 nt, 2 chunks) ----
  for (int cb = 0; cb < 40; cb += 20) {
    v8f a2[20];
    if (active) {
#pragma unroll
      for (int j = 0; j < 20; ++j) {
        float bv = b2[(cb + j) * 16 + c];
        a2[j] = (v8f){bv, bv, bv, bv, bv, bv, bv, bv};
      }
    }
    stage_slice(pw2 + (size_t)cb * 512, wbuf[0], 20 * 64, tid);
    stage_sync();
    for (int kk = 0; kk < 8; ++kk) {
      if (kk + 1 < 8)
        stage_slice(pw2 + (size_t)((kk + 1) * 40 + cb) * 512, wbuf[(kk + 1) & 1],
                    20 * 64, tid);
      if (active)
        gemm_ktile<20>(&curh[wid][c * 256], kk, g, lane, wbuf[kk & 1], a2);
      stage_sync();
    }
    if (active) {
#pragma unroll
      for (int j = 0; j < 20; ++j) {
        int nt = cb + j;
#pragma unroll
        for (int r = 0; r < 8; ++r) {
          float v = fmaxf(a2[j][r], 0.0f);
          int m = r + 8 * g;
          int col = nt * 16 + c;
          if (col < H_N) {
            atomicAdd(&pooled[(size_t)sSh[wid][m] * H_N + col], v);
          } else if (col < H_N + DOUT_N) {
            outP[(size_t)(t0 + m) * DOUT_N + (col - H_N)] = v;
          } else {
            atomicAdd(&pooled[(size_t)oSh[wid][m] * H_N + (col - H_N - DOUT_N)], v);
          }
        }
      }
    }
  }
}

// ---------- fused per-object MLP ----------
__global__ __launch_bounds__(256) void obj_mlp(
    const float* __restrict__ pooled, const float* __restrict__ counts,
    const unsigned short* __restrict__ pw3, const float* __restrict__ b3,
    const unsigned short* __restrict__ pw4, const float* __restrict__ b4,
    float* __restrict__ outObj)
{
  __shared__ __align__(16) unsigned short ph[8][16 * 256];
  __shared__ __align__(16) unsigned short h2[8][16 * 256];
  __shared__ __align__(16) unsigned short wb[2][16 * 512];

  const int tid  = threadIdx.x;
  const int wid  = tid >> 5;
  const int lane = tid & 31;
  const int g = lane >> 4, c = lane & 15;
  const int o0 = blockIdx.x * 128 + wid * 16;
  const bool active = (o0 < O_N);

  // normalize pooled rows and convert to bf16 LDS [16][256]
  if (active) {
    for (int m = 0; m < 16; ++m) {
      int oi = o0 + m;
      float inv = 1.0f / fmaxf(counts[oi], 1.0f);
#pragma unroll
      for (int j = 0; j < 2; ++j) {
        float4 v = *(const float4*)(pooled + (size_t)oi * H_N + j * 128 + lane * 4);
        unsigned lo = (unsigned)f2bf(v.x * inv) | ((unsigned)f2bf(v.y * inv) << 16);
        unsigned hi = (unsigned)f2bf(v.z * inv) | ((unsigned)f2bf(v.w * inv) << 16);
        *(uint2*)&ph[wid][m * 256 + j * 128 + lane * 4] = make_uint2(lo, hi);
      }
    }
  }

  // h2 = relu(pn @ W3 + b3) ; K=256 (8 ktiles), N=256 (16 ntiles)
  v8f a3[16];
  if (active) {
#pragma unroll
    for (int nt = 0; nt < 16; ++nt) {
      float bv = b3[nt * 16 + c];
      a3[nt] = (v8f){bv, bv, bv, bv, bv, bv, bv, bv};
    }
  }
  stage_slice(pw3, wb[0], 16 * 64, tid);
  stage_sync();
  for (int kk = 0; kk < 8; ++kk) {
    if (kk + 1 < 8)
      stage_slice(pw3 + (size_t)((kk + 1) * 16) * 512, wb[(kk + 1) & 1], 16 * 64, tid);
    if (active)
      gemm_ktile<16>(&ph[wid][c * 256], kk, g, lane, wb[kk & 1], a3);
    stage_sync();
  }
  if (active) {
#pragma unroll
    for (int nt = 0; nt < 16; ++nt) {
#pragma unroll
      for (int r = 0; r < 8; ++r) {
        int m = r + 8 * g;
        h2[wid][m * 256 + nt * 16 + c] = f2bf(fmaxf(a3[nt][r], 0.0f));
      }
    }
  }

  // new_obj = relu(h2 @ W4 + b4) ; K=256 (8 ktiles), N=128 (8 ntiles)
  v8f a4[8];
  if (active) {
#pragma unroll
    for (int nt = 0; nt < 8; ++nt) {
      float bv = b4[nt * 16 + c];
      a4[nt] = (v8f){bv, bv, bv, bv, bv, bv, bv, bv};
    }
  }
  stage_slice(pw4, wb[0], 8 * 64, tid);
  stage_sync();
  for (int kk = 0; kk < 8; ++kk) {
    if (kk + 1 < 8)
      stage_slice(pw4 + (size_t)((kk + 1) * 8) * 512, wb[(kk + 1) & 1], 8 * 64, tid);
    if (active)
      gemm_ktile<8>(&h2[wid][c * 256], kk, g, lane, wb[kk & 1], a4);
    stage_sync();
  }
  if (active) {
#pragma unroll
    for (int nt = 0; nt < 8; ++nt) {
#pragma unroll
      for (int r = 0; r < 8; ++r) {
        int m = r + 8 * g;
        outObj[(size_t)(o0 + m) * DOUT_N + nt * 16 + c] = fmaxf(a4[nt][r], 0.0f);
      }
    }
  }
}

// ---------- launch ----------

extern "C" void kernel_launch(void* const* d_in, const int* in_sizes, int n_in,
                              void* d_out, int out_size, void* d_ws, size_t ws_size,
                              hipStream_t stream) {
  const float* obj   = (const float*)d_in[0];
  const float* pred  = (const float*)d_in[1];
  const int*   edges = (const int*)d_in[2];
  const float* W1 = (const float*)d_in[3];
  const float* b1 = (const float*)d_in[4];
  const float* W2 = (const float*)d_in[5];
  const float* b2 = (const float*)d_in[6];
  const float* W3 = (const float*)d_in[7];
  const float* b3 = (const float*)d_in[8];
  const float* W4 = (const float*)d_in[9];
  const float* b4 = (const float*)d_in[10];

  char* ws = (char*)d_ws;
  float* pooled = (float*)ws;                 ws += (size_t)O_N * H_N * 4;   // 51.2 MB
  float* counts = (float*)ws;                 ws += (size_t)O_N * 4;         // 0.2 MB
  unsigned short* pw1 = (unsigned short*)ws;  ws += (size_t)384 * 256 * 2;
  unsigned short* pw2 = (unsigned short*)ws;  ws += (size_t)256 * 640 * 2;
  unsigned short* pw3 = (unsigned short*)ws;  ws += (size_t)256 * 256 * 2;
  unsigned short* pw4 = (unsigned short*)ws;  ws += (size_t)256 * 128 * 2;

  float* outObj = (float*)d_out;
  float* outP   = (float*)d_out + (size_t)O_N * DOUT_N;

  zero_f32<<<2048, 256, 0, stream>>>(pooled, (long)O_N * H_N + O_N);

  pack_w<<<(384 * 256 + 255) / 256, 256, 0, stream>>>(W1, pw1, 384, 256);
  pack_w<<<(256 * 640 + 255) / 256, 256, 0, stream>>>(W2, pw2, 256, 640);
  pack_w<<<(256 * 256 + 255) / 256, 256, 0, stream>>>(W3, pw3, 256, 256);
  pack_w<<<(256 * 128 + 255) / 256, 256, 0, stream>>>(W4, pw4, 256, 128);

  count_edges<<<(T_N + 255) / 256, 256, 0, stream>>>(edges, counts);

  triple_mlp<<<(T_N + 127) / 128, 256, 0, stream>>>(obj, pred, edges,
                                                    pw1, b1, pw2, b2, pooled, outP);
  obj_mlp<<<(O_N + 127) / 128, 256, 0, stream>>>(pooled, counts, pw3, b3, pw4, b4,
                                                 outObj);
}
